// BoneCloud_81046032875683
// MI455X (gfx1250) — matrix operand order repaired and checked by reading, not compiled, not run on hardware
//
#include <hip/hip_runtime.h>
#include <hip/hip_bf16.h>

#define N_POINTS 200000
#define N_BONES  512
#define SIGMA    20.0f

typedef __attribute__((ext_vector_type(16))) _Float16 v16h;
typedef __attribute__((ext_vector_type(8)))  _Float16 v8h;
typedef __attribute__((ext_vector_type(8)))  float    v8f;
typedef __attribute__((ext_vector_type(2)))  float    v2f;

// ---------------------------------------------------------------------------
// Prep: build (a) bonesaug[512][4] = (bx,by,bz,||b||^2) for the f32 dist-WMMA
// B operand, and (b) tpack = the [512 x 16] f16 transform matrix pre-swizzled
// into per-lane WMMA-B register order for v_wmma_f32_16x16x32_f16:
//   flat index t = (superchunk*32 + lane)*16 + k  ->  Tmat[32*sc + 16*(lane>=16) + k][lane%16]
// Tmat columns 0..11 = 3x4 transform rows, col 12 = 1.0 (softmax denominator), 13..15 = 0.
// ---------------------------------------------------------------------------
__global__ void bonecloud_prep(const float* __restrict__ bone_locs,
                               const float* __restrict__ bone_transf,
                               const int* __restrict__ tidx_p,
                               float* __restrict__ bonesaug,
                               _Float16* __restrict__ tpack) {
    int t = blockIdx.x * blockDim.x + threadIdx.x;   // 0 .. 8191
    if (t >= N_BONES * 16) return;
    int tidx = tidx_p[0];

    int k  = t & 15;            // K index within 16-row group
    int l  = (t >> 4) & 31;     // lane
    int sc = t >> 9;            // superchunk (32 bones each)
    int n  = l & 15;            // column of Tmat
    int j  = sc * 32 + ((l >> 4) << 4) + k;   // bone index

    const float* p = bone_transf + ((size_t)tidx * N_BONES + j) * 9;
    // 6D -> rotmat (Gram-Schmidt), rotcont layout: a1 = (p0,p2,p4), a2 = (p1,p3,p5)
    float a1x = p[0], a1y = p[2], a1z = p[4];
    float a2x = p[1], a2y = p[3], a2z = p[5];
    float inv1 = 1.0f / (sqrtf(a1x*a1x + a1y*a1y + a1z*a1z) + 1e-12f);
    float b1x = a1x*inv1, b1y = a1y*inv1, b1z = a1z*inv1;
    float dd  = b1x*a2x + b1y*a2y + b1z*a2z;
    float px = a2x - dd*b1x, py = a2y - dd*b1y, pz = a2z - dd*b1z;
    float inv2 = 1.0f / (sqrtf(px*px + py*py + pz*pz) + 1e-12f);
    float b2x = px*inv2, b2y = py*inv2, b2z = pz*inv2;
    float b3x = b1y*b2z - b1z*b2y;
    float b3y = b1z*b2x - b1x*b2z;
    float b3z = b1x*b2y - b1y*b2x;
    // R = [b1 b2 b3] as columns; transform row i = [R[i][0..2], transl[i]]
    float R[3][4] = {
        { b1x, b2x, b3x, p[6] },
        { b1y, b2y, b3y, p[7] },
        { b1z, b2z, b3z, p[8] },
    };
    float val;
    if (n < 12)       val = R[n >> 2][n & 3];
    else if (n == 12) val = 1.0f;            // accumulates softmax denominator
    else              val = 0.0f;
    tpack[t] = (_Float16)val;

    if (t < N_BONES) {
        float bx = bone_locs[3*t+0], by = bone_locs[3*t+1], bz = bone_locs[3*t+2];
        bonesaug[4*t+0] = bx;
        bonesaug[4*t+1] = by;
        bonesaug[4*t+2] = bz;
        bonesaug[4*t+3] = bx*bx + by*by + bz*bz;
    }
}

// ---------------------------------------------------------------------------
// Main: 256 threads = 8 waves per block; each wave owns a 16-point tile.
// Per 32-bone superchunk:
//   2x v_wmma_f32_16x16x4_f32  -> (||b||^2 - 2 x.b) tiles
//   VALU: +||x||^2, sqrt, exp2 (scaled softmax numerator), cvt f16 -> LDS
//   re-load LDS in A-layout, 1x v_wmma_f32_16x16x32_f16 accumulating
//   [weights] x [transforms | ones] into a 16x16 f32 C tile.
// Final: y = (C[m,0:12] . xh) / C[m,12].
// ---------------------------------------------------------------------------
__global__ void __launch_bounds__(256)
bonecloud_skin(const float* __restrict__ xyz,
               const float* __restrict__ bonesaug,
               const _Float16* __restrict__ tpack,
               float* __restrict__ out) {
    __shared__ float s_xsq[8][16];
    __shared__ __align__(16) _Float16 s_e[8][16][32];   // per-wave weight tile
    __shared__ float s_c[8][16][16];                    // per-wave result dump

    const int lane = threadIdx.x & 31;
    const int wave = threadIdx.x >> 5;
    const int hi   = lane >> 4;       // 0: lanes 0-15, 1: lanes 16-31
    const int ml   = lane & 15;
    const int base = blockIdx.x * 128 + wave * 16;

    int pidx = base + ml;
    if (pidx > N_POINTS - 1) pidx = N_POINTS - 1;       // clamp; store is guarded
    float x0 = xyz[3*pidx+0], x1 = xyz[3*pidx+1], x2 = xyz[3*pidx+2];
    float xsq = x0*x0 + x1*x1 + x2*x2;
    s_xsq[wave][ml] = xsq;
    __syncthreads();

    float xsqr[8];
#pragma unroll
    for (int r = 0; r < 8; ++r) xsqr[r] = s_xsq[wave][r + 8*hi];

    // A operand of dist WMMA (16x4 f32): row m = (-2x,-2y,-2z,1)
    v2f av;
    av.x = hi ? (-2.0f * x2) : (-2.0f * x0);
    av.y = hi ? 1.0f         : (-2.0f * x1);

    v8f acc = {};   // blend accumulator (16x16 f32 C tile)
    const float kExpScale = -SIGMA * 1.44269504088896340736f;  // -sigma*log2(e)

    for (int cc = 0; cc < 16; ++cc) {          // 16 superchunks of 32 bones
#pragma unroll
        for (int half = 0; half < 2; ++half) {
            int j = (cc*2 + half)*16 + ml;     // bone for this lane's B column
            v2f bv = *(const v2f*)(bonesaug + j*4 + 2*hi);
            v8f cz = {};
            v8f d = __builtin_amdgcn_wmma_f32_16x16x4_f32(
                false, av, false, bv, (short)0, cz, false, false);
#pragma unroll
            for (int r = 0; r < 8; ++r) {
                float d2   = fmaxf(d[r] + xsqr[r], 0.0f);
                float dist = __builtin_amdgcn_sqrtf(d2);
                // exp(-sigma*dist) * 2^14 : constant scale cancels in normalization,
                // keeps f16 weights out of the denormal range.
                float e = __builtin_amdgcn_exp2f(fmaf(dist, kExpScale, 14.0f));
                s_e[wave][r + 8*hi][half*16 + ml] = (_Float16)e;
            }
        }
        asm volatile("" ::: "memory");   // keep DS store->load program order

        // A-layout load for f16 WMMA: lane<16 -> row ml, K 0-15;
        // lane>=16 -> row ml, K 16-31.
        int k0 = 16*hi;
        v8h lo  = *(const v8h*)&s_e[wave][ml][k0];
        v8h hi8 = *(const v8h*)&s_e[wave][ml][8 + k0];
        v16h aE = __builtin_shufflevector(lo, hi8,
                    0,1,2,3,4,5,6,7,8,9,10,11,12,13,14,15);
        // B operand pre-swizzled: one 32B vector load per lane
        v16h bT = *(const v16h*)(tpack + (cc*32 + lane)*16);
        acc = __builtin_amdgcn_wmma_f32_16x16x32_f16(
            false, aE, false, bT, (short)0, acc, false, false);
        asm volatile("" ::: "memory");
    }

    // Dump C tile and finish: y_i = (row[4i..4i+2].x + row[4i+3]) / S
#pragma unroll
    for (int r = 0; r < 8; ++r) s_c[wave][r + 8*hi][ml] = acc[r];
    asm volatile("" ::: "memory");

    if (lane < 16) {
        int p = base + lane;
        if (p < N_POINTS) {
            const float* row = s_c[wave][lane];
            float invS = 1.0f / row[12];
            float y0 = (row[0]*x0 + row[1]*x1 + row[2]*x2  + row[3])  * invS;
            float y1 = (row[4]*x0 + row[5]*x1 + row[6]*x2  + row[7])  * invS;
            float y2 = (row[8]*x0 + row[9]*x1 + row[10]*x2 + row[11]) * invS;
            out[3*p+0] = y0;
            out[3*p+1] = y1;
            out[3*p+2] = y2;
        }
    }
}

extern "C" void kernel_launch(void* const* d_in, const int* in_sizes, int n_in,
                              void* d_out, int out_size, void* d_ws, size_t ws_size,
                              hipStream_t stream) {
    const float* xyz         = (const float*)d_in[0];
    const float* bone_locs   = (const float*)d_in[1];
    const float* bone_transf = (const float*)d_in[2];
    const int*   tidx        = (const int*)d_in[3];
    float* out = (float*)d_out;

    float*    bonesaug = (float*)d_ws;                                  // 8 KB
    _Float16* tpack    = (_Float16*)((char*)d_ws + N_BONES*4*sizeof(float)); // 16 KB

    hipLaunchKernelGGL(bonecloud_prep, dim3(32), dim3(256), 0, stream,
                       bone_locs, bone_transf, tidx, bonesaug, tpack);

    int nblocks = (N_POINTS + 127) / 128;   // 128 points per block (8 waves x 16)
    hipLaunchKernelGGL(bonecloud_skin, dim3(nblocks), dim3(256), 0, stream,
                       xyz, bonesaug, tpack, out);
}
